// BiLSTM_CRF_86294482911655
// MI455X (gfx1250) — matrix operand order repaired and checked by reading, not compiled, not run on hardware
//
#include <hip/hip_runtime.h>

// ---------------------------------------------------------------------------
// BiLSTM-CRF for MI455X (gfx1250, wave32).
//   S=8192 seq, E=768 embed, H=768 hidden, T=7 tags.
// Phase 1: xp = x @ w_ih^T + b_ih  (both dirs) -> bf16 WMMA GEMM, TDM-staged A.
// Phase 2: sequential LSTM recurrence, 32 WGs/direction, w_hh slice in LDS,
//          device-scope per-step barrier (atomic arrive + spin).
// Phase 3: tag projection (wave per row), Phase 4: CRF forward scan (1 wave).
// ---------------------------------------------------------------------------

typedef __bf16 bf16;
typedef bf16  v16bf __attribute__((ext_vector_type(16)));
typedef bf16  v8bf  __attribute__((ext_vector_type(8)));
typedef float v8f   __attribute__((ext_vector_type(8)));
typedef unsigned int v4u __attribute__((ext_vector_type(4)));
typedef int   v4i   __attribute__((ext_vector_type(4)));
typedef int   v8i   __attribute__((ext_vector_type(8)));

constexpr int S   = 8192;
constexpr int E   = 768;
constexpr int H   = 768;
constexpr int G4H = 4 * H;   // 3072
constexpr int T   = 7;

// ---------------- workspace layout (bytes) ----------------
constexpr size_t XP_ELEMS    = (size_t)S * G4H;           // 25.2M floats
constexpr size_t XP_F_OFF    = 0;
constexpr size_t XP_B_OFF    = XP_F_OFF + XP_ELEMS * 4;
constexpr size_t HS_ELEMS    = (size_t)S * 2 * H;
constexpr size_t HS_OFF      = XP_B_OFF + XP_ELEMS * 4;
constexpr size_t FEATS_OFF   = HS_OFF + HS_ELEMS * 4;     // S x 8 padded
constexpr size_t XBF_OFF     = FEATS_OFF + (size_t)S * 8 * 4;
constexpr size_t WP_ELEMS    = (size_t)G4H * E;
constexpr size_t WPF_OFF     = XBF_OFF + (size_t)S * E * 2;
constexpr size_t WPB_OFF     = WPF_OFF + WP_ELEMS * 2;
constexpr size_t HBUF_OFF    = WPB_OFF + WP_ELEMS * 2;    // [2 dir][2 bufs][H] f32
constexpr size_t BAR_OFF     = HBUF_OFF + 2 * 2 * H * 4;  // 2 u32 counters

// ---------------------------------------------------------------------------
// Kernel 0: cast x -> bf16 row-major; pack w_ih into B-fragment-friendly
// layout Wpack[kchunk][col][32] so a lane's 16-element B fragment is one
// contiguous 32B load.
// ---------------------------------------------------------------------------
__global__ void pack_kernel(const float* __restrict__ x,
                            const float* __restrict__ w_ih_f,
                            const float* __restrict__ w_ih_b,
                            bf16* __restrict__ Xbf,
                            bf16* __restrict__ WpF,
                            bf16* __restrict__ WpB)
{
    size_t idx = (size_t)blockIdx.x * blockDim.x + threadIdx.x;
    if (idx < (size_t)S * E) Xbf[idx] = (bf16)x[idx];
    if (idx < (size_t)G4H * E) {
        int o = (int)(idx / E), e = (int)(idx % E);
        size_t dst = ((size_t)(e >> 5) * G4H + o) * 32 + (e & 31);
        WpF[dst] = (bf16)w_ih_f[idx];
        WpB[dst] = (bf16)w_ih_b[idx];
    }
}

// ---------------------------------------------------------------------------
// Kernel 1: zero barrier counters, seed h_buf with h0 (determinism: every call)
// ---------------------------------------------------------------------------
__global__ void init_kernel(const float* __restrict__ h0,
                            float* __restrict__ h_buf,
                            unsigned* __restrict__ bar)
{
    int i = blockIdx.x * blockDim.x + threadIdx.x;
    if (i < 2) bar[i] = 0u;
    if (i < 2 * H) {
        int dir = i / H, j = i % H;
        h_buf[dir * 2 * H + j] = h0[dir * H + j];   // buffer (t&1)==0
    }
}

// ---------------------------------------------------------------------------
// Kernel 2: XP = Xbf(SxE) * Wpack + bias  (f32 out), both directions (grid.z).
// Block = 128 thr (4 waves). Block tile: 16 rows x 256 cols; wave: 16x64
// (4 accumulators). K-loop: 24 chunks of 32. A tile (16x768 bf16, 24 KB)
// staged once into LDS by the Tensor Data Mover.
// ---------------------------------------------------------------------------
__launch_bounds__(128)
__global__ void xp_gemm_kernel(const bf16* __restrict__ Xbf,
                               const bf16* __restrict__ WpF,
                               const bf16* __restrict__ WpB,
                               const float* __restrict__ b_ih_f,
                               const float* __restrict__ b_ih_b,
                               float* __restrict__ xp_f,
                               float* __restrict__ xp_b)
{
    __shared__ bf16 lds_x[16 * E];   // 24 KB

    const int dir = blockIdx.z;
    const bf16*  Wp   = dir ? WpB : WpF;
    const float* bias = dir ? b_ih_b : b_ih_f;
    float*       XP   = dir ? xp_b : xp_f;

    const int s0     = blockIdx.x * 16;
    const int wave   = threadIdx.x >> 5;
    const int lane   = threadIdx.x & 31;
    const int o_wave = blockIdx.y * 256 + wave * 64;

#if defined(__gfx1250__) && __has_builtin(__builtin_amdgcn_tensor_load_to_lds)
    if (wave == 0) {
        // --- Tensor DMA descriptor (D#): 2D tensor ExS, tile 768x16, 2B elems
        unsigned long long ga = (unsigned long long)(uintptr_t)(Xbf + (size_t)s0 * E);
        unsigned lds_addr = (unsigned)(uintptr_t)(void*)lds_x;
        v4u g0;
        g0.x = 1u;                                            // count=1 valid
        g0.y = lds_addr;                                      // lds_addr
        g0.z = (unsigned)(ga & 0xFFFFFFFFu);                  // global_addr lo
        g0.w = (unsigned)((ga >> 32) & 0x01FFFFFFu) | (2u << 30); // hi | type=2
        v8i g1;
        g1[0] = (int)(1u << 16);                              // data_size=2B
        g1[1] = (int)(((unsigned)E & 0xFFFFu) << 16);         // tensor_dim0 lo
        g1[2] = (int)((((unsigned)E >> 16) & 0xFFFFu) |
                      (((unsigned)S & 0xFFFFu) << 16));       // dim0 hi|dim1 lo
        g1[3] = (int)((((unsigned)S >> 16) & 0xFFFFu) |
                      (((unsigned)E & 0xFFFFu) << 16));       // dim1 hi|tile_d0
        g1[4] = (int)16u;                                     // tile_dim1=16
        g1[5] = (int)E;                                       // dim0_stride lo
        g1[6] = 0;
        g1[7] = 0;
        v4i gz4 = {0, 0, 0, 0};
        v8i gz8 = {0, 0, 0, 0, 0, 0, 0, 0};
        // amdgpu-toolchain (clang-23) 6-arg form:
        //   (v4u group0, v8i group1, v4i group2, v4i group3, v8i extra, i32 cpol)
        __builtin_amdgcn_tensor_load_to_lds(g0, g1, gz4, gz4, gz8, 0);
        __builtin_amdgcn_s_wait_tensorcnt(0);
    }
    __syncthreads();
#else
    for (int i = threadIdx.x; i < 16 * E / 8; i += blockDim.x)
        ((v8bf*)lds_x)[i] = ((const v8bf*)(Xbf + (size_t)s0 * E))[i];
    __syncthreads();
#endif

    v8f acc0 = {}, acc1 = {}, acc2 = {}, acc3 = {};
    const int m    = lane & 15;
    const int koff = (lane >> 4) * 16;   // A frag: lanes 16-31 hold K=16..31
    const int n    = lane & 15;
    const int kb   = (lane >> 4) * 16;   // B frag: lanes 16-31 hold K=16..31

    for (int c = 0; c < E / 32; ++c) {
        // A fragment (16x32 bf16), contiguous k = 16*(lane>>4) + s, same order as B
        const bf16* ap = lds_x + m * E + c * 32 + koff;
        v8bf alo = *(const v8bf*)(ap);
        v8bf ahi = *(const v8bf*)(ap + 8);
        v16bf a;
#pragma unroll
        for (int j = 0; j < 8; ++j) { a[j] = alo[j]; a[8 + j] = ahi[j]; }

        const bf16* bb = Wp + (size_t)c * G4H * 32 + kb;
        v16bf b0 = *(const v16bf*)(bb + (size_t)(o_wave +  0 + n) * 32);
        v16bf b1 = *(const v16bf*)(bb + (size_t)(o_wave + 16 + n) * 32);
        v16bf b2 = *(const v16bf*)(bb + (size_t)(o_wave + 32 + n) * 32);
        v16bf b3 = *(const v16bf*)(bb + (size_t)(o_wave + 48 + n) * 32);

        acc0 = __builtin_amdgcn_wmma_f32_16x16x32_bf16(false, a, false, b0, (short)0, acc0, false, false);
        acc1 = __builtin_amdgcn_wmma_f32_16x16x32_bf16(false, a, false, b1, (short)0, acc1, false, false);
        acc2 = __builtin_amdgcn_wmma_f32_16x16x32_bf16(false, a, false, b2, (short)0, acc2, false, false);
        acc3 = __builtin_amdgcn_wmma_f32_16x16x32_bf16(false, a, false, b3, (short)0, acc3, false, false);
    }

    // C/D layout: VGPR r -> row r (lanes 0-15) / r+8 (lanes 16-31), N = lane&15
    const int mbase = (lane >> 4) * 8;
#pragma unroll
    for (int r = 0; r < 8; ++r) {
        int row = s0 + r + mbase;
        float* orow = XP + (size_t)row * G4H + o_wave + n;
        orow[0]  = acc0[r] + bias[o_wave +  0 + n];
        orow[16] = acc1[r] + bias[o_wave + 16 + n];
        orow[32] = acc2[r] + bias[o_wave + 32 + n];
        orow[48] = acc3[r] + bias[o_wave + 48 + n];
    }
}

// ---------------------------------------------------------------------------
// Kernel 3: persistent LSTM recurrence. 64 blocks: 0-31 forward, 32-63
// backward. Each block pins its 24 hidden units' 96 gate rows (294 KB) in
// LDS; per step each wave computes 3 units x 4 gates via lane-strided dots +
// wave32 shuffle reduction; inter-block sync = monotone atomic counter.
// ---------------------------------------------------------------------------
__launch_bounds__(256)
__global__ void lstm_rec_kernel(const float* __restrict__ xp_f,
                                const float* __restrict__ xp_b,
                                const float* __restrict__ w_hh_f,
                                const float* __restrict__ w_hh_b,
                                const float* __restrict__ b_hh_f,
                                const float* __restrict__ b_hh_b,
                                const float* __restrict__ c0,
                                float* __restrict__ h_buf,   // [2][2][H]
                                unsigned* __restrict__ bar,  // [2]
                                float* __restrict__ hs)      // [S][2H]
{
    extern __shared__ float smem[];
    float* wsl   = smem;             // [96][H]  294,912 B
    float* h_lds = smem + 96 * H;    // [H]

    const int dir = blockIdx.x >> 5;
    const int blk = blockIdx.x & 31;
    const int u0  = blk * 24;
    const float* XP  = dir ? xp_b : xp_f;
    const float* Whh = dir ? w_hh_b : w_hh_f;
    const float* Bhh = dir ? b_hh_b : b_hh_f;

    for (int row = 0; row < 96; ++row) {            // row = lu*4 + gate
        int g = row & 3, lu = row >> 2;
        const float* src = Whh + (size_t)(g * H + u0 + lu) * H;
        for (int e = threadIdx.x; e < H; e += 256)
            wsl[row * H + e] = src[e];
    }
    __syncthreads();

    const int wave = threadIdx.x >> 5, lane = threadIdx.x & 31;
    float c_st[3] = {0.f, 0.f, 0.f};
    if (lane == 0) {
#pragma unroll
        for (int q = 0; q < 3; ++q)
            c_st[q] = c0[dir * H + u0 + wave * 3 + q];
    }
    float* hcur0 = h_buf + dir * 2 * H;

    for (int t = 0; t < S; ++t) {
        const int s = dir ? (S - 1 - t) : t;
        const float* hprev = hcur0 + (t & 1) * H;
        float* hnext = hcur0 + ((t + 1) & 1) * H;
        for (int e = threadIdx.x; e < H; e += 256) h_lds[e] = hprev[e];
        __syncthreads();

#pragma unroll
        for (int q = 0; q < 3; ++q) {
            const int lu = wave * 3 + q;
            const float* w0 = wsl + (lu * 4 + 0) * H;
            const float* w1 = wsl + (lu * 4 + 1) * H;
            const float* w2 = wsl + (lu * 4 + 2) * H;
            const float* w3 = wsl + (lu * 4 + 3) * H;
            float a0 = 0.f, a1 = 0.f, a2 = 0.f, a3 = 0.f;
            for (int e = lane; e < H; e += 32) {
                float hv = h_lds[e];
                a0 = fmaf(w0[e], hv, a0);
                a1 = fmaf(w1[e], hv, a1);
                a2 = fmaf(w2[e], hv, a2);
                a3 = fmaf(w3[e], hv, a3);
            }
#pragma unroll
            for (int off = 16; off > 0; off >>= 1) {
                a0 += __shfl_down(a0, off, 32);
                a1 += __shfl_down(a1, off, 32);
                a2 += __shfl_down(a2, off, 32);
                a3 += __shfl_down(a3, off, 32);
            }
            if (lane == 0) {
                const int j  = u0 + lu;
                const size_t rb = (size_t)s * G4H;
                float gi = XP[rb + 0 * H + j] + Bhh[0 * H + j] + a0;
                float gf = XP[rb + 1 * H + j] + Bhh[1 * H + j] + a1;
                float gg = XP[rb + 2 * H + j] + Bhh[2 * H + j] + a2;
                float go = XP[rb + 3 * H + j] + Bhh[3 * H + j] + a3;
                float i_ = 1.f / (1.f + __expf(-gi));
                float f_ = 1.f / (1.f + __expf(-gf));
                float g_ = tanhf(gg);
                float o_ = 1.f / (1.f + __expf(-go));
                float c_ = f_ * c_st[q] + i_ * g_;
                c_st[q] = c_;
                float h_ = o_ * tanhf(c_);
                hnext[j] = h_;
                hs[(size_t)s * (2 * H) + dir * H + j] = h_;
            }
        }
        __threadfence();
        __syncthreads();
        if (threadIdx.x == 0) {
            atomicAdd(&bar[dir], 1u);
            unsigned target = 32u * (unsigned)(t + 1);
            while (__hip_atomic_load(&bar[dir], __ATOMIC_ACQUIRE,
                                     __HIP_MEMORY_SCOPE_AGENT) < target)
                __builtin_amdgcn_s_sleep(1);
        }
        __syncthreads();
    }
}

// ---------------------------------------------------------------------------
// Kernel 4: feats = hs @ w_tag^T + b_tag. One wave per row, 7 tag accums.
// ---------------------------------------------------------------------------
__launch_bounds__(256)
__global__ void tag_proj_kernel(const float* __restrict__ hs,
                                const float* __restrict__ w_tag,
                                const float* __restrict__ b_tag,
                                float* __restrict__ feats)
{
    const int wave = threadIdx.x >> 5, lane = threadIdx.x & 31;
    const int row  = blockIdx.x * 8 + wave;
    if (row >= S) return;
    const float* hrow = hs + (size_t)row * (2 * H);
    float acc[T] = {0, 0, 0, 0, 0, 0, 0};
    for (int e = lane; e < 2 * H; e += 32) {
        float hv = hrow[e];
#pragma unroll
        for (int tg = 0; tg < T; ++tg)
            acc[tg] = fmaf(hv, w_tag[tg * (2 * H) + e], acc[tg]);
    }
#pragma unroll
    for (int tg = 0; tg < T; ++tg)
#pragma unroll
        for (int off = 16; off > 0; off >>= 1)
            acc[tg] += __shfl_down(acc[tg], off, 32);
    if (lane == 0)
#pragma unroll
        for (int tg = 0; tg < T; ++tg)
            feats[(size_t)row * 8 + tg] = acc[tg] + b_tag[tg];
}

// ---------------------------------------------------------------------------
// Kernel 5: CRF forward scan, single wave32. Lane j (<7) owns alpha[j].
// ---------------------------------------------------------------------------
__launch_bounds__(32)
__global__ void crf_kernel(const float* __restrict__ feats,
                           const float* __restrict__ transitions,
                           const float* __restrict__ start_trans,
                           const float* __restrict__ end_trans,
                           float* __restrict__ out)
{
    const int j = threadIdx.x;
    const bool act = j < T;
    float tr[T];
#pragma unroll
    for (int i = 0; i < T; ++i) tr[i] = act ? transitions[i * T + j] : -1e30f;
    float a = act ? (start_trans[j] + feats[j]) : -1e30f;
    for (int t = 1; t < S; ++t) {
        float emit = act ? feats[(size_t)t * 8 + j] : 0.f;
        float m = -1e30f, v[T];
#pragma unroll
        for (int i = 0; i < T; ++i) {
            v[i] = __shfl(a, i, 32) + tr[i];
            m = fmaxf(m, v[i]);
        }
        float ssum = 0.f;
#pragma unroll
        for (int i = 0; i < T; ++i) ssum += __expf(v[i] - m);
        a = act ? (m + __logf(ssum) + emit) : -1e30f;
    }
    float z = act ? a + end_trans[j] : -1e30f;
    float m = z;
#pragma unroll
    for (int off = 16; off > 0; off >>= 1) m = fmaxf(m, __shfl_xor(m, off, 32));
    float e = __expf(z - m);
#pragma unroll
    for (int off = 16; off > 0; off >>= 1) e += __shfl_xor(e, off, 32);
    if (j == 0) out[0] = m + __logf(e);
}

// ---------------------------------------------------------------------------
extern "C" void kernel_launch(void* const* d_in, const int* in_sizes, int n_in,
                              void* d_out, int out_size, void* d_ws, size_t ws_size,
                              hipStream_t stream)
{
    const float* x       = (const float*)d_in[0];
    const float* w_ih_f  = (const float*)d_in[1];
    const float* w_hh_f  = (const float*)d_in[2];
    const float* b_ih_f  = (const float*)d_in[3];
    const float* b_hh_f  = (const float*)d_in[4];
    const float* w_ih_b  = (const float*)d_in[5];
    const float* w_hh_b  = (const float*)d_in[6];
    const float* b_ih_b  = (const float*)d_in[7];
    const float* b_hh_b  = (const float*)d_in[8];
    const float* h0      = (const float*)d_in[9];
    const float* c0      = (const float*)d_in[10];
    const float* w_tag   = (const float*)d_in[11];
    const float* b_tag   = (const float*)d_in[12];
    const float* trans   = (const float*)d_in[13];
    const float* st_tr   = (const float*)d_in[14];
    const float* en_tr   = (const float*)d_in[15];

    char* ws = (char*)d_ws;
    float*    xp_f  = (float*)(ws + XP_F_OFF);
    float*    xp_b  = (float*)(ws + XP_B_OFF);
    float*    hs    = (float*)(ws + HS_OFF);
    float*    feats = (float*)(ws + FEATS_OFF);
    bf16*     Xbf   = (bf16*)(ws + XBF_OFF);
    bf16*     WpF   = (bf16*)(ws + WPF_OFF);
    bf16*     WpB   = (bf16*)(ws + WPB_OFF);
    float*    h_buf = (float*)(ws + HBUF_OFF);
    unsigned* bar   = (unsigned*)(ws + BAR_OFF);

    pack_kernel<<<dim3(((size_t)S * E + 255) / 256), 256, 0, stream>>>(
        x, w_ih_f, w_ih_b, Xbf, WpF, WpB);
    init_kernel<<<dim3(8), 256, 0, stream>>>(h0, h_buf, bar);
    xp_gemm_kernel<<<dim3(S / 16, G4H / 256, 2), 128, 0, stream>>>(
        Xbf, WpF, WpB, b_ih_f, b_ih_b, xp_f, xp_b);
    const size_t lstm_lds = (96 * H + H) * sizeof(float);  // ~298 KB of 320 KB
    lstm_rec_kernel<<<dim3(64), 256, lstm_lds, stream>>>(
        xp_f, xp_b, w_hh_f, w_hh_b, b_hh_f, b_hh_b, c0, h_buf, bar, hs);
    tag_proj_kernel<<<dim3(S / 8), 256, 0, stream>>>(hs, w_tag, b_tag, feats);
    crf_kernel<<<dim3(1), 32, 0, stream>>>(feats, trans, st_tr, en_tr,
                                           (float*)d_out);
}
